// FastenNet_43679817401126
// MI455X (gfx1250) — compile-verified, matrix operand order checked by reading, not run on hardware
//
#include <hip/hip_runtime.h>
#include <math.h>

// ---------------------------------------------------------------------------
// R-GCN (2 layers) for MI455X / gfx1250 (wave32, WMMA).
// Memory-bound (~400MB traffic, ~1.3 GFLOP -> ~17us HBM roofline): gather /
// scatter kernels carry the traffic; the einsum('nh,rhc->rnc') runs on
// v_wmma_f32_16x16x32_f16 with two relations packed per 16-wide B tile (C=8).
// N_NODES % 16 == 0, so every 16-row WMMA tile is fully valid or fully out of
// range -> tail predication is one wave-uniform early exit, stores are
// straight-line (no exec-mask dance).
// ---------------------------------------------------------------------------

typedef _Float16 half_t;
typedef __attribute__((ext_vector_type(16))) _Float16 v16h;
typedef __attribute__((ext_vector_type(8)))  _Float16 v8h;
typedef __attribute__((ext_vector_type(4)))  _Float16 v4h;
typedef __attribute__((ext_vector_type(8)))  float    v8f;

#define N_NODES 20000
#define R_REL   32
#define H_DIM   64
#define C_CLS   8
#define E_EDGES 640000
#define NPAIR   17            // 16 relation pairs + 1 "root" pair (root2 | 0)
#define LDS_STRIDE 72         // halves per row (64 + pad); 144B row pitch

// ---------------- kernel 0: zero the h accumulator (float4) ----------------
__global__ void k_zero(float4* __restrict__ p, int n4) {
  int i = blockIdx.x * blockDim.x + threadIdx.x;
  if (i < n4) p[i] = make_float4(0.f, 0.f, 0.f, 0.f);
}

// ---------------- kernel 1: conv1 gather + scatter-add ----------------
// msg1[e] = W1[edge_type[e], src[e]] (64 f32), h[dst[e]] += msg1[e].
// 16 threads per edge, each moves a float4 (b128 load, 4 f32 atomics).
__global__ void k_conv1_scatter(const int* __restrict__ edge_index,
                                const int* __restrict__ edge_type,
                                const float* __restrict__ W1,
                                float* __restrict__ h) {
  long long t = (long long)blockIdx.x * blockDim.x + threadIdx.x;
  int e = (int)(t >> 4);
  if (e >= E_EDGES) return;
  int j = ((int)t & 15) * 4;
  int s = edge_index[e];
  int d = edge_index[E_EDGES + e];
  int r = edge_type[e];
  const float* w = W1 + ((long long)r * N_NODES + s) * H_DIM + j;
  float4 v = *(const float4*)w;
  float* hp = h + (long long)d * H_DIM + j;
  atomicAdd(hp + 0, v.x);
  atomicAdd(hp + 1, v.y);
  atomicAdd(hp + 2, v.z);
  atomicAdd(hp + 3, v.w);
}

// ---------------- kernel 2: h = relu(h + root1 + bias1), float4 ----------------
__global__ void k_bias_relu(float4* __restrict__ h,
                            const float4* __restrict__ root1,
                            const float4* __restrict__ bias1) {
  int i = blockIdx.x * blockDim.x + threadIdx.x;
  if (i >= (N_NODES * H_DIM) / 4) return;
  float4 a = h[i];
  float4 b = root1[i];
  float4 c = bias1[i & ((H_DIM / 4) - 1)];
  float4 o;
  o.x = fmaxf(a.x + b.x + c.x, 0.f);
  o.y = fmaxf(a.y + b.y + c.y, 0.f);
  o.z = fmaxf(a.z + b.z + c.z, 0.f);
  o.w = fmaxf(a.w + b.w + c.w, 0.f);
  h[i] = o;
}

// ---------------- kernel 3: pack B operands in WMMA register order ----------
// B tile for pair p, k-step s is a 32x16 (KxN) f16 matrix:
//   columns 0..7  -> classes of relation 2p, columns 8..15 -> relation 2p+1
//   (pair 16: columns 0..7 -> root2, columns 8..15 -> zeros)
// Wave32 B layout: lane holds col = lane%16; halves j<8 are K = kb+j,
// halves j>=8 are K = kb+16+(j-8), kb = (lane<16 ? 0 : 8).
// Layout: Bpack[((p*2+s)*32 + lane)*16 + j]  -> per-lane v16h = 2 b128 loads.
__global__ void k_pack_b(const float* __restrict__ W2,
                         const float* __restrict__ root2,
                         half_t* __restrict__ Bpack) {
  const int total = NPAIR * 2 * 32 * 16;
  for (int i = threadIdx.x; i < total; i += blockDim.x) {
    int j    = i & 15;
    int lane = (i >> 4) & 31;
    int s    = (i >> 9) & 1;
    int p    = i >> 10;
    int col  = lane & 15;
    int kb   = (lane < 16) ? 0 : 8;
    int k32  = (j < 8) ? (kb + j) : (kb + 16 + (j - 8));
    int K    = s * 32 + k32;
    int c    = col & 7;
    float v;
    if (p < 16) {
      int rel = 2 * p + (col >> 3);
      v = W2[((long long)rel * H_DIM + K) * C_CLS + c];
    } else {
      v = (col < 8) ? root2[K * C_CLS + c] : 0.0f;
    }
    Bpack[i] = (half_t)v;
  }
}

// ---------------- kernel 4: WMMA einsum + out initialization ----------------
// Each block owns 128 node rows (8 waves x 16-row tiles). The h tile is staged
// to LDS once as f16 (float4 loads -> packed b64 LDS stores); each wave builds
// its A fragments once and reuses them across all 17 B pairs. Two chained
// wmma_f32_16x16x32_f16 cover K=64 per tile.
__global__ void __launch_bounds__(256)
k_wmma_einsum(const float* __restrict__ h,
              const half_t* __restrict__ Bpack,
              const float* __restrict__ bias2,
              float* __restrict__ h_rel,
              float* __restrict__ out) {
  __shared__ half_t lds[128 * LDS_STRIDE];
  const int tid = threadIdx.x;
  const int rowBase = blockIdx.x * 128;

  // stage h tile (f32 -> f16), 2048 float4 chunks; clamp tail-row reads
  for (int i = tid; i < 128 * (H_DIM / 4); i += 256) {
    int r  = i >> 4;             // 16 float4 chunks per 64-wide row
    int c4 = (i & 15) * 4;
    int gr = rowBase + r;
    if (gr >= N_NODES) gr = N_NODES - 1;
    float4 v = *(const float4*)(h + (long long)gr * H_DIM + c4);
    v4h hv = { (half_t)v.x, (half_t)v.y, (half_t)v.z, (half_t)v.w };
    *(v4h*)(&lds[r * LDS_STRIDE + c4]) = hv;    // 8B aligned (144B row pitch)
  }
  __syncthreads();

  const int wave = tid >> 5, lane = tid & 31;
  const int tileRow = rowBase + (wave << 4);
  if (tileRow >= N_NODES) return;  // N%16==0: tile fully valid or fully out

  const int arow = (wave << 4) + (lane & 15);
  const int kb   = (lane < 16) ? 0 : 8;
  const half_t* abase = &lds[arow * LDS_STRIDE];

  // A fragments: 16-bit A 16x32 layout -> per-lane halves {kb..kb+7, kb+16..kb+23}
  v8h a0lo = *(const v8h*)(abase + 0 * 32 + kb);
  v8h a0hi = *(const v8h*)(abase + 0 * 32 + kb + 16);
  v8h a1lo = *(const v8h*)(abase + 1 * 32 + kb);
  v8h a1hi = *(const v8h*)(abase + 1 * 32 + kb + 16);
  v16h a0 = __builtin_shufflevector(a0lo, a0hi, 0,1,2,3,4,5,6,7,8,9,10,11,12,13,14,15);
  v16h a1 = __builtin_shufflevector(a1lo, a1hi, 0,1,2,3,4,5,6,7,8,9,10,11,12,13,14,15);

  const int col  = lane & 15;
  const int c    = col & 7;
  const int row0 = tileRow + ((lane >= 16) ? 8 : 0);

  // ---- 16 relation pairs -> h_rel ----
  for (int p = 0; p < 16; ++p) {
    v16h b0 = *(const v16h*)(Bpack + ((p * 2 + 0) * 32 + lane) * 16);
    v16h b1 = *(const v16h*)(Bpack + ((p * 2 + 1) * 32 + lane) * 16);
    v8f acc = {};
    acc = __builtin_amdgcn_wmma_f32_16x16x32_f16(false, a0, false, b0,
                                                 (short)0, acc, false, false);
    acc = __builtin_amdgcn_wmma_f32_16x16x32_f16(false, a1, false, b1,
                                                 (short)0, acc, false, false);
    // C/D layout: VGPR v at lane l = element (M = v + (l>=16?8:0), N = l%16)
    int rel = 2 * p + (col >> 3);
    float* dst = h_rel + ((long long)rel * N_NODES + row0) * C_CLS + c;
#pragma unroll
    for (int v = 0; v < 8; ++v) dst[v * C_CLS] = acc[v];
  }

  // ---- root pair -> out = h @ root2 + bias2 (columns 8..15 are zero-padded) ----
  {
    v16h b0 = *(const v16h*)(Bpack + ((16 * 2 + 0) * 32 + lane) * 16);
    v16h b1 = *(const v16h*)(Bpack + ((16 * 2 + 1) * 32 + lane) * 16);
    v8f acc = {};
    acc = __builtin_amdgcn_wmma_f32_16x16x32_f16(false, a0, false, b0,
                                                 (short)0, acc, false, false);
    acc = __builtin_amdgcn_wmma_f32_16x16x32_f16(false, a1, false, b1,
                                                 (short)0, acc, false, false);
    if (col < 8) {
      float bb = bias2[col];
      float* dst = out + (long long)row0 * C_CLS + col;
#pragma unroll
      for (int v = 0; v < 8; ++v) dst[v * C_CLS] = acc[v] + bb;
    }
  }
}

// ---------------- kernel 5: conv2 gather + scatter-add ----------------
// out[dst[e]] += h_rel[edge_type[e], src[e]]; 8 threads per edge (one class).
__global__ void k_conv2_scatter(const int* __restrict__ edge_index,
                                const int* __restrict__ edge_type,
                                const float* __restrict__ h_rel,
                                float* __restrict__ out) {
  long long t = (long long)blockIdx.x * blockDim.x + threadIdx.x;
  int e = (int)(t >> 3);
  if (e >= E_EDGES) return;
  int c = (int)t & 7;
  int s = edge_index[e];
  int d = edge_index[E_EDGES + e];
  int r = edge_type[e];
  float v = h_rel[((long long)r * N_NODES + s) * C_CLS + c];
  atomicAdd(&out[(long long)d * C_CLS + c], v);
}

// ---------------- kernel 6: row-wise log_softmax over 8 classes ----------------
__global__ void k_log_softmax(float* __restrict__ out) {
  int n = blockIdx.x * blockDim.x + threadIdx.x;
  if (n >= N_NODES) return;
  float x[C_CLS];
  float4 lo = *(const float4*)(out + (long long)n * C_CLS);
  float4 hi = *(const float4*)(out + (long long)n * C_CLS + 4);
  x[0] = lo.x; x[1] = lo.y; x[2] = lo.z; x[3] = lo.w;
  x[4] = hi.x; x[5] = hi.y; x[6] = hi.z; x[7] = hi.w;
  float m = x[0];
#pragma unroll
  for (int i = 1; i < C_CLS; ++i) m = fmaxf(m, x[i]);
  float s = 0.0f;
#pragma unroll
  for (int i = 0; i < C_CLS; ++i) s += expf(x[i] - m);
  float lg = m + logf(s);
  float4 o0 = { x[0] - lg, x[1] - lg, x[2] - lg, x[3] - lg };
  float4 o1 = { x[4] - lg, x[5] - lg, x[6] - lg, x[7] - lg };
  *(float4*)(out + (long long)n * C_CLS)     = o0;
  *(float4*)(out + (long long)n * C_CLS + 4) = o1;
}

extern "C" void kernel_launch(void* const* d_in, const int* in_sizes, int n_in,
                              void* d_out, int out_size, void* d_ws, size_t ws_size,
                              hipStream_t stream) {
  (void)in_sizes; (void)n_in; (void)out_size; (void)ws_size;
  const int*   edge_index = (const int*)d_in[0];     // [2, E]
  const int*   edge_type  = (const int*)d_in[1];     // [E]
  // d_in[2] = tensor_slice (unused; edge_type is read directly per edge)
  const float* W1    = (const float*)d_in[3];        // [R, N, H]
  const float* root1 = (const float*)d_in[4];        // [N, H]
  const float* bias1 = (const float*)d_in[5];        // [H]
  const float* W2    = (const float*)d_in[6];        // [R, H, C]
  const float* root2 = (const float*)d_in[7];        // [H, C]
  const float* bias2 = (const float*)d_in[8];        // [C]
  float* out = (float*)d_out;                        // [N, C]

  // workspace layout
  char*   ws    = (char*)d_ws;
  float*  h     = (float*)ws;                                          // N*H f32
  float*  h_rel = (float*)(ws + (size_t)N_NODES * H_DIM * 4);          // R*N*C f32
  half_t* Bpack = (half_t*)(ws + (size_t)N_NODES * H_DIM * 4
                               + (size_t)R_REL * N_NODES * C_CLS * 4); // packed B

  k_zero<<<(N_NODES * H_DIM / 4 + 255) / 256, 256, 0, stream>>>(
      (float4*)h, N_NODES * H_DIM / 4);
  k_conv1_scatter<<<(E_EDGES * 16) / 256, 256, 0, stream>>>(
      edge_index, edge_type, W1, h);
  k_bias_relu<<<(N_NODES * H_DIM / 4 + 255) / 256, 256, 0, stream>>>(
      (float4*)h, (const float4*)root1, (const float4*)bias1);
  k_pack_b<<<1, 256, 0, stream>>>(W2, root2, Bpack);
  k_wmma_einsum<<<(N_NODES + 127) / 128, 256, 0, stream>>>(
      h, Bpack, bias2, h_rel, out);
  k_conv2_scatter<<<(E_EDGES * 8) / 256, 256, 0, stream>>>(
      edge_index, edge_type, h_rel, out);
  k_log_softmax<<<(N_NODES + 255) / 256, 256, 0, stream>>>(out);
}
